// point_transformer_29832842838690
// MI455X (gfx1250) — matrix operand order repaired and checked by reading, not compiled, not run on hardware
//
#include <hip/hip_runtime.h>
#include <hip/hip_bf16.h>
#include <math.h>

typedef __attribute__((ext_vector_type(16))) _Float16 v16h;
typedef __attribute__((ext_vector_type(8)))  float    v8f;

#define F_IN   16
#define DIM    32
#define ATTN_H 16
#define POS_H  32
#define NPTS   1024
#define NB     2
#define NTOT   (NB * NPTS)            // 2048 points
#define NEG_INF (-3.0e38f)

// ---------------------------------------------------------------------------
// helpers
// ---------------------------------------------------------------------------
__device__ __forceinline__ v8f wmma16(v16h a, v16h b, v8f c) {
  // D = A(16x32 f16) x B(32x16 f16) + C(16x16 f32)
  return __builtin_amdgcn_wmma_f32_16x16x32_f16(false, a, false, b,
                                                (short)0, c, false, false);
}

// relu for ALU-derived (already-canonical) values: single v_max from compiler
__device__ __forceinline__ float relu1(float x) { return fmaxf(x, 0.0f); }

// relu for WMMA results: avoid the compiler's extra canonicalize op
__device__ __forceinline__ float reluw(float x) {
  float r;
  asm("v_max_num_f32 %0, 0, %1" : "=v"(r) : "v"(x));
  return r;
}

__device__ __forceinline__ v8f bcast8(float x) {
  v8f c;
#pragma unroll
  for (int r = 0; r < 8; ++r) c[r] = x;
  return c;
}

// A-fragment k index for element e of v16h (16-bit A 16x32 layout)
__device__ __forceinline__ int kA(int e, int g) {
  return (e & 7) + 8 * g + ((e >= 8) ? 16 : 0);
}

// ---------------------------------------------------------------------------
// Phase 1a: f = relu(feature @ W1 + b1)    [NTOT, DIM]
// ---------------------------------------------------------------------------
__global__ void pt_lin1_kernel(const float* __restrict__ feat,
                               const float* __restrict__ W1,
                               const float* __restrict__ b1,
                               float* __restrict__ f) {
  int idx = blockIdx.x * blockDim.x + threadIdx.x;
  if (idx >= NTOT * DIM) return;
  int n = idx >> 5, d = idx & 31;
  const float* fr = feat + (size_t)n * F_IN;
  float acc = b1[d];
#pragma unroll
  for (int c = 0; c < F_IN; ++c) acc += fr[c] * W1[c * DIM + d];
  f[idx] = relu1(acc);
}

// ---------------------------------------------------------------------------
// Phase 1b: q,k,v = relu(f @ W{q,k,v} + b)   each [NTOT, DIM]
// ---------------------------------------------------------------------------
__global__ void pt_qkv_kernel(const float* __restrict__ f,
                              const float* __restrict__ Wq, const float* __restrict__ bq,
                              const float* __restrict__ Wk, const float* __restrict__ bk,
                              const float* __restrict__ Wv, const float* __restrict__ bv,
                              float* __restrict__ q, float* __restrict__ k,
                              float* __restrict__ v) {
  int idx = blockIdx.x * blockDim.x + threadIdx.x;
  if (idx >= NTOT * DIM) return;
  int n = idx >> 5, d = idx & 31;
  const float* fr = f + (size_t)n * DIM;
  float aq = bq[d], ak = bk[d], av = bv[d];
#pragma unroll
  for (int c = 0; c < DIM; ++c) {
    float x = fr[c];
    aq += x * Wq[c * DIM + d];
    ak += x * Wk[c * DIM + d];
    av += x * Wv[c * DIM + d];
  }
  q[idx] = relu1(aq);
  k[idx] = relu1(ak);
  v[idx] = relu1(av);
}

// ---------------------------------------------------------------------------
// Phase 1c: pp = pos @ Wp1  [NTOT, POS_H]   (linear part of pos layer 1)
// ---------------------------------------------------------------------------
__global__ void pt_pp_kernel(const float* __restrict__ pos,
                             const float* __restrict__ Wp1,
                             float* __restrict__ pp) {
  int idx = blockIdx.x * blockDim.x + threadIdx.x;
  if (idx >= NTOT * POS_H) return;
  int n = idx >> 5, d = idx & 31;
  const float* pr = pos + (size_t)n * 3;
  pp[idx] = pr[0] * Wp1[0 * POS_H + d] + pr[1] * Wp1[1 * POS_H + d] +
            pr[2] * Wp1[2 * POS_H + d];
}

// ---------------------------------------------------------------------------
// Phase 1d: qa = q @ Wa1  [NTOT, ATTN_H]   (linear part of (qk+pe)@Wa1)
// ---------------------------------------------------------------------------
__global__ void pt_qa_kernel(const float* __restrict__ q,
                             const float* __restrict__ Wa1,
                             float* __restrict__ qa) {
  int idx = blockIdx.x * blockDim.x + threadIdx.x;
  if (idx >= NTOT * ATTN_H) return;
  int n = idx >> 4, d = idx & 15;
  const float* qr = q + (size_t)n * DIM;
  float acc = 0.0f;
#pragma unroll
  for (int c = 0; c < DIM; ++c) acc += qr[c] * Wa1[c * ATTN_H + d];
  qa[idx] = acc;
}

// ---------------------------------------------------------------------------
// Phase 1e: mask -> additive float bias (0 / -1e9)
// ---------------------------------------------------------------------------
__global__ void pt_smask_kernel(const unsigned char* __restrict__ mask,
                                float* __restrict__ sm) {
  int idx = blockIdx.x * blockDim.x + threadIdx.x;
  if (idx < NTOT) sm[idx] = mask[idx] ? 0.0f : -1.0e9f;
}

// ---------------------------------------------------------------------------
// Phase 2: fused pairwise MLPs + per-channel online softmax + output linear.
// One wave32 per output point i; 4 waves per block; 2 j-tiles (32 rows) per
// iteration. pos layer1 / qk are linearized; the pos layer-2 A-fragment is
// built directly from pp rows (no LDS round trip); WMMA-result relus use a
// single-instruction asm max (no canonicalize).
// ---------------------------------------------------------------------------
__global__ __launch_bounds__(128)
void pt_attn_kernel(const float* __restrict__ pos,
                    const float* __restrict__ pp,     // pos @ Wp1
                    const float* __restrict__ qa,     // q @ Wa1
                    const float* __restrict__ kmat,
                    const float* __restrict__ vmat,
                    const float* __restrict__ smaskf,
                    const float* __restrict__ Wp1, const float* __restrict__ bp1,
                    const float* __restrict__ Wp2, const float* __restrict__ bp2,
                    const float* __restrict__ Wa1, const float* __restrict__ ba1,
                    const float* __restrict__ Wa2, const float* __restrict__ ba2,
                    const float* __restrict__ W2,  const float* __restrict__ b2,
                    float* __restrict__ out) {
  __shared__ float stage_a1[4][2][16][16];  // a1 D->A staging
  __shared__ float stage_pe[4][2][16][32];  // pos_emb D->A staging
  __shared__ float kvec[4][32];             // key vector of row i
  __shared__ float ovec[4][32];             // init: hb staging; end: output vec

  const int tid  = threadIdx.x;
  const int wv   = tid >> 5;            // wave id in block
  const int lane = tid & 31;
  const int ln   = lane & 15;           // column / row-within-16
  const int g    = (lane >> 4) & 1;     // half-wave group

  const int ig = blockIdx.x * 4 + wv;   // global point index
  const int b  = ig >> 10;
  const int i  = ig & (NPTS - 1);

  // per-wave base pointers with batch folded in (int32 offsets below)
  const float* ppb = pp     + (size_t)b * NPTS * POS_H;
  const float* qab = qa     + (size_t)b * NPTS * ATTN_H;
  const float* vb  = vmat   + (size_t)b * NPTS * DIM;
  const float* smb = smaskf + b * NPTS;

  // ---- B fragments (f16), k(e) = kA(e, g), col = ln ----
  v16h Bp2_lo, Bp2_hi, Ba1f, Ba2_lo, Ba2_hi;
#pragma unroll
  for (int e = 0; e < 16; ++e) {
    int k = kA(e, g);
    Bp2_lo[e] = (_Float16)(Wp2[k * DIM + ln]);
    Bp2_hi[e] = (_Float16)(Wp2[k * DIM + ln + 16]);
    Ba1f[e]   = (_Float16)(Wa1[k * ATTN_H + ln]);
    Ba2_lo[e] = (_Float16)((k < 16) ? Wa2[k * DIM + ln]      : 0.0f);
    Ba2_hi[e] = (_Float16)((k < 16) ? Wa2[k * DIM + ln + 16] : 0.0f);
  }
  const v8f Cbp2_lo = bcast8(bp2[ln]);
  const v8f Cbp2_hi = bcast8(bp2[ln + 16]);
  const v8f Cba2_lo = bcast8(ba2[ln]);
  const v8f Cba2_hi = bcast8(ba2[ln + 16]);

  // ---- per-i data ----
  kvec[wv][lane] = kmat[(size_t)(b * NPTS + i) * DIM + lane];
  const float pix = pos[(size_t)(b * NPTS + i) * 3 + 0];
  const float piy = pos[(size_t)(b * NPTS + i) * 3 + 1];
  const float piz = pos[(size_t)(b * NPTS + i) * 3 + 2];
  // h column bias: bp1[n] - pos_i @ Wp1[:,n]  (staged to LDS, then gathered
  // per-lane in A-fragment element order -> loop-invariant hbk[16])
  ovec[wv][ln] = bp1[ln] -
      (pix * Wp1[0 * POS_H + ln] + piy * Wp1[1 * POS_H + ln] +
       piz * Wp1[2 * POS_H + ln]);
  ovec[wv][ln + 16] = bp1[ln + 16] -
      (pix * Wp1[0 * POS_H + ln + 16] + piy * Wp1[1 * POS_H + ln + 16] +
       piz * Wp1[2 * POS_H + ln + 16]);
  // softmax is invariant to a row-uniform shift: (m_i & m_j ? 0 : -1e9)
  // is equivalent (post-softmax) to smj * (m_i ? 1 : 0).
  const float mflag = (smb[i] == 0.0f) ? 1.0f : 0.0f;
  __builtin_amdgcn_wave_barrier();
  float hbk[16];
#pragma unroll
  for (int e = 0; e < 16; ++e) hbk[e] = ovec[wv][kA(e, g)];
  // attn layer-1 column bias: ba1[n] - k_i @ Wa1[:,n]
  float cb1 = ba1[ln];
#pragma unroll
  for (int c = 0; c < DIM; ++c) cb1 -= kvec[wv][c] * Wa1[c * ATTN_H + ln];
  __builtin_amdgcn_wave_barrier();

  // online-softmax state: channel d0 = ln, channel d1 = ln+16
  float m0 = NEG_INF, s0 = 0.0f, ac0 = 0.0f;
  float m1 = NEG_INF, s1 = 0.0f, ac1 = 0.0f;

#pragma unroll 1
  for (int jt = 0; jt < NPTS / 32; ++jt) {
    const int j0 = jt * 32;

    // ===== stage 1+2: A = relu(pp[j] + hb) built directly (no LDS trip),
    //                  then pos_emb = relu(A @ Wp2 + bp2) =====
    v16h Ah[2];
#pragma unroll
    for (int u = 0; u < 2; ++u) {
      const int j = j0 + 16 * u + ln;
      const float4* pr4 = (const float4*)(ppb + j * POS_H);
      float pv[16];
      *(float4*)&pv[0]  = pr4[2 * g];      // k =  8g .. 8g+3
      *(float4*)&pv[4]  = pr4[2 * g + 1];  // k =  8g+4 .. 8g+7
      *(float4*)&pv[8]  = pr4[4 + 2 * g];  // k = 16+8g ..
      *(float4*)&pv[12] = pr4[5 + 2 * g];
#pragma unroll
      for (int e = 0; e < 16; ++e)
        Ah[u][e] = (_Float16)relu1(pv[e] + hbk[e]);
    }

    v8f pe0[2], pe1[2];
#pragma unroll
    for (int u = 0; u < 2; ++u) {
      pe0[u] = wmma16(Ah[u], Bp2_lo, Cbp2_lo);
      pe1[u] = wmma16(Ah[u], Bp2_hi, Cbp2_hi);
    }
#pragma unroll
    for (int u = 0; u < 2; ++u)
#pragma unroll
      for (int r = 0; r < 8; ++r) {
        pe0[u][r] = reluw(pe0[u][r]);
        pe1[u][r] = reluw(pe1[u][r]);
        stage_pe[wv][u][r + 8 * g][ln]      = pe0[u][r];
        stage_pe[wv][u][r + 8 * g][ln + 16] = pe1[u][r];
      }
    __builtin_amdgcn_wave_barrier();

    // ===== stage 3: a1 = relu(pe @ Wa1 + (qa[j] - k_i@Wa1 + ba1)) =====
    v16h Ape[2];
    v8f  C3[2];
#pragma unroll
    for (int u = 0; u < 2; ++u) {
#pragma unroll
      for (int e = 0; e < 16; ++e)
        Ape[u][e] = (_Float16)stage_pe[wv][u][ln][kA(e, g)];
#pragma unroll
      for (int r = 0; r < 8; ++r) {
        const int j = j0 + 16 * u + r + 8 * g;
        C3[u][r] = qab[j * ATTN_H + ln] + cb1;
      }
    }
    v8f a1d[2];
#pragma unroll
    for (int u = 0; u < 2; ++u) a1d[u] = wmma16(Ape[u], Ba1f, C3[u]);
#pragma unroll
    for (int u = 0; u < 2; ++u)
#pragma unroll
      for (int r = 0; r < 8; ++r)
        stage_a1[wv][u][r + 8 * g][ln] = reluw(a1d[u][r]);
    __builtin_amdgcn_wave_barrier();

    // ===== stage 4: r = relu(a1 @ Wa2 + ba2) =====
    v16h Aa[2];
#pragma unroll
    for (int u = 0; u < 2; ++u)
#pragma unroll
      for (int e = 0; e < 16; ++e) {
        int k = kA(e, g);
        Aa[u][e] = (k < 16) ? (_Float16)stage_a1[wv][u][ln][k] : (_Float16)0.0f;
      }
    v8f r0[2], r1[2];
#pragma unroll
    for (int u = 0; u < 2; ++u) {
      r0[u] = wmma16(Aa[u], Ba2_lo, Cba2_lo);
      r1[u] = wmma16(Aa[u], Ba2_hi, Cba2_hi);
    }

    // ===== stage 5: logits + online softmax =====
#pragma unroll
    for (int u = 0; u < 2; ++u) {
      float l0[8], l1[8], vj0[8], vj1[8];
      float t0 = NEG_INF, t1 = NEG_INF;
#pragma unroll
      for (int r = 0; r < 8; ++r) {
        const int j = j0 + 16 * u + r + 8 * g;
        const float smj = smb[j] * mflag;
        l0[r] = reluw(r0[u][r]) + pe0[u][r] + smj;
        l1[r] = reluw(r1[u][r]) + pe1[u][r] + smj;
        vj0[r] = vb[j * DIM + ln]      + pe0[u][r];
        vj1[r] = vb[j * DIM + ln + 16] + pe1[u][r];
        t0 = fmaxf(t0, l0[r]);
        t1 = fmaxf(t1, l1[r]);
      }
      t0 = fmaxf(t0, __shfl_xor(t0, 16, 32));
      t1 = fmaxf(t1, __shfl_xor(t1, 16, 32));
      const float nm0 = fmaxf(m0, t0), nm1 = fmaxf(m1, t1);
      const float sc0 = __expf(m0 - nm0), sc1 = __expf(m1 - nm1);
      float ps0 = 0.0f, pa0 = 0.0f, ps1 = 0.0f, pa1 = 0.0f;
#pragma unroll
      for (int r = 0; r < 8; ++r) {
        const float p0 = __expf(l0[r] - nm0);
        const float p1 = __expf(l1[r] - nm1);
        ps0 += p0; pa0 += p0 * vj0[r];
        ps1 += p1; pa1 += p1 * vj1[r];
      }
      ps0 += __shfl_xor(ps0, 16, 32);
      pa0 += __shfl_xor(pa0, 16, 32);
      ps1 += __shfl_xor(ps1, 16, 32);
      pa1 += __shfl_xor(pa1, 16, 32);
      s0 = s0 * sc0 + ps0; ac0 = ac0 * sc0 + pa0; m0 = nm0;
      s1 = s1 * sc1 + ps1; ac1 = ac1 * sc1 + pa1; m1 = nm1;
    }
  }

  // ---- finalize: o = acc/s, then out = o @ W2 + b2 ----
  if (lane < 16) {
    ovec[wv][ln]      = ac0 / s0;
    ovec[wv][ln + 16] = ac1 / s1;
  }
  __builtin_amdgcn_wave_barrier();

  float y = b2[lane];
#pragma unroll
  for (int d = 0; d < DIM; ++d) y += ovec[wv][d] * W2[d * DIM + lane];
  out[(size_t)(b * NPTS + i) * DIM + lane] = y;
}

// ---------------------------------------------------------------------------
// Launch
// ---------------------------------------------------------------------------
extern "C" void kernel_launch(void* const* d_in, const int* in_sizes, int n_in,
                              void* d_out, int out_size, void* d_ws, size_t ws_size,
                              hipStream_t stream) {
  const float* feature = (const float*)d_in[0];
  const float* pos     = (const float*)d_in[1];
  const unsigned char* mask = (const unsigned char*)d_in[2];
  const float* W1 = (const float*)d_in[3];   const float* b1 = (const float*)d_in[4];
  const float* Wq = (const float*)d_in[5];   const float* bq = (const float*)d_in[6];
  const float* Wk = (const float*)d_in[7];   const float* bk = (const float*)d_in[8];
  const float* Wv = (const float*)d_in[9];   const float* bv = (const float*)d_in[10];
  const float* Wp1 = (const float*)d_in[11]; const float* bp1 = (const float*)d_in[12];
  const float* Wp2 = (const float*)d_in[13]; const float* bp2 = (const float*)d_in[14];
  const float* Wa1 = (const float*)d_in[15]; const float* ba1 = (const float*)d_in[16];
  const float* Wa2 = (const float*)d_in[17]; const float* ba2 = (const float*)d_in[18];
  const float* W2 = (const float*)d_in[19];  const float* b2 = (const float*)d_in[20];

  float* ws = (float*)d_ws;
  float* f  = ws;                          // NTOT*DIM
  float* q  = ws + 1 * NTOT * DIM;
  float* k  = ws + 2 * NTOT * DIM;
  float* v  = ws + 3 * NTOT * DIM;
  float* pp = ws + 4 * NTOT * DIM;         // NTOT*POS_H
  float* qa = ws + 5 * NTOT * DIM;         // NTOT*ATTN_H
  float* sm = ws + 5 * NTOT * DIM + NTOT * ATTN_H;  // NTOT

  pt_lin1_kernel<<<(NTOT * DIM + 255) / 256, 256, 0, stream>>>(feature, W1, b1, f);
  pt_qkv_kernel<<<(NTOT * DIM + 255) / 256, 256, 0, stream>>>(
      f, Wq, bq, Wk, bk, Wv, bv, q, k, v);
  pt_pp_kernel<<<(NTOT * POS_H + 255) / 256, 256, 0, stream>>>(pos, Wp1, pp);
  pt_qa_kernel<<<(NTOT * ATTN_H + 255) / 256, 256, 0, stream>>>(q, Wa1, qa);
  pt_smask_kernel<<<(NTOT + 255) / 256, 256, 0, stream>>>(mask, sm);

  pt_attn_kernel<<<NTOT / 4, 128, 0, stream>>>(
      pos, pp, qa, k, v, sm,
      Wp1, bp1, Wp2, bp2, Wa1, ba1, Wa2, ba2, W2, b2,
      (float*)d_out);
}
